// VGAE_893353197865
// MI455X (gfx1250) — matrix operand order repaired and checked by reading, not compiled, not run on hardware
//
#include <hip/hip_runtime.h>
#include <math.h>

// ---------------------------------------------------------------------------
// VGAE forward on gfx1250 (wave32, WMMA).
//   Feature GEMMs -> v_wmma_f32_16x16x32_bf16 with bf16x3 split (fp32-class
//                    accuracy for the mu/logstd outputs).
//   z@z^T         -> z pre-packed to bf16 (2 MB, L2-resident), packed b128
//                    fragment loads, 16x64 register tile per wave, sigmoid
//                    epilogue. 256 MB adj store is the HBM floor (~11us).
//   GCN aggregation -> global_atomic_add_f32 at L2.
// ---------------------------------------------------------------------------

typedef __attribute__((ext_vector_type(16))) __bf16        v16bf;
typedef __attribute__((ext_vector_type(8)))  float         v8f;
typedef __attribute__((ext_vector_type(4)))  unsigned int  v4u;

#define KN    8192
#define KIN   512
#define KHID  256
#define KLAT  128
#define KE    262144

__device__ __forceinline__ unsigned short vgae_bf16_rne(float f) {
  unsigned int u = __float_as_uint(f);
  u += 0x7FFFu + ((u >> 16) & 1u);           // round to nearest even
  return (unsigned short)(u >> 16);
}
__device__ __forceinline__ float vgae_bf16_hi(unsigned short h) {
  return __uint_as_float(((unsigned int)h) << 16);
}

union VgaeBfFrag { v16bf v; v4u q[2]; unsigned short u[16]; };

// ------------------------- degree / norm kernels ---------------------------

__global__ void vgae_fill1(float* __restrict__ p, int n) {
  int i = blockIdx.x * blockDim.x + threadIdx.x;
  if (i < n) p[i] = 1.0f;                    // self-loop contribution
}

__global__ void vgae_deg_edges(const int* __restrict__ dst, float* __restrict__ deg) {
  int e = blockIdx.x * blockDim.x + threadIdx.x;
  if (e < KE) atomicAdd(&deg[dst[e]], 1.0f);
}

__global__ void vgae_deg_to_dinv(float* __restrict__ p, int n) {
  int i = blockIdx.x * blockDim.x + threadIdx.x;
  if (i < n) { float d = p[i]; p[i] = (d > 0.0f) ? rsqrtf(d) : 0.0f; }
}

// ------------------------- GEMM: C = A[MxK] * B[KxN] -----------------------
// One 16x16 output tile per wave, bf16x3 split (hi*hi + hi*lo + lo*hi).
// Fragment layouts per ISA 7.12.2 (wave32):
//  A 16x32 bf16: lane(0..15)=row M; K = 16*(v>>2) + 8*half + 2*(v&3) + j
//  B 32x16 bf16: lane(0..15)=col N; K = 16*(v>>2) + 8*half + 2*(v&3) + j
//  C 16x16 f32:  lane(0..15)=col N; row = vgpr + 8*half

__global__ void vgae_gemm_bf16x3(const float* __restrict__ A, const float* __restrict__ B,
                                 float* __restrict__ C, int M, int N, int K) {
  const int lane  = threadIdx.x & 31;
  const int laneL = lane & 15;
  const int half  = lane >> 4;
  const int t     = blockIdx.x * (blockDim.x >> 5) + (threadIdx.x >> 5);
  const int numMt = M >> 4;
  const int mt = t % numMt, nt = t / numMt;
  const int m0 = mt << 4, n0 = nt << 4;

  v8f acc = {};
  for (int kk = 0; kk < K; kk += 32) {
    VgaeBfFrag ahi, alo, bhi, blo;
#pragma unroll
    for (int v = 0; v < 8; ++v) {
      const int ka = kk + ((v >> 2) << 4) + (half << 3) + ((v & 3) << 1);
#pragma unroll
      for (int j = 0; j < 2; ++j) {
        float xv = A[(size_t)(m0 + laneL) * K + (ka + j)];
        unsigned short h = vgae_bf16_rne(xv);
        ahi.u[2 * v + j] = h;
        alo.u[2 * v + j] = vgae_bf16_rne(xv - vgae_bf16_hi(h));
      }
      const int kb = kk + ((v >> 2) << 4) + (half << 3) + ((v & 3) << 1);
#pragma unroll
      for (int j = 0; j < 2; ++j) {
        float wv = B[(size_t)(kb + j) * N + (n0 + laneL)];
        unsigned short h = vgae_bf16_rne(wv);
        bhi.u[2 * v + j] = h;
        blo.u[2 * v + j] = vgae_bf16_rne(wv - vgae_bf16_hi(h));
      }
    }
    acc = __builtin_amdgcn_wmma_f32_16x16x32_bf16(false, alo.v, false, bhi.v,
                                                  (short)0, acc, false, false);
    acc = __builtin_amdgcn_wmma_f32_16x16x32_bf16(false, ahi.v, false, blo.v,
                                                  (short)0, acc, false, false);
    acc = __builtin_amdgcn_wmma_f32_16x16x32_bf16(false, ahi.v, false, bhi.v,
                                                  (short)0, acc, false, false);
  }
#pragma unroll
  for (int r = 0; r < 8; ++r)
    C[(size_t)(m0 + (half << 3) + r) * N + (n0 + laneL)] = acc[r];
}

// ------------------ adj = sigmoid(Zb * Zb^T), Zb: bf16 [8192 x 128] --------
// Pre-packed bf16 -> fragment loads are contiguous 16B chunks per lane:
//   A-frag: row m0+laneL, K chunk  [kk+16h .. +15]
//   B-frag (col n): row n of Zb,  K chunk  [kk+16h .. +15]
// Each wave computes a 16x64 strip: 1 A-frag reused across 4 B-frags.

__device__ __forceinline__ void vgae_store_sig(const v8f& acc, float* __restrict__ adj,
                                               int m0, int col, int laneL, int half) {
#pragma unroll
  for (int r = 0; r < 8; ++r) {
    float s = 1.0f / (1.0f + __expf(-acc[r]));
    adj[(size_t)(m0 + (half << 3) + r) * KN + (col + laneL)] = s;
  }
}

__global__ void vgae_zzt_sigmoid(const unsigned short* __restrict__ Zb,
                                 float* __restrict__ adj) {
  const int lane  = threadIdx.x & 31;
  const int laneL = lane & 15;
  const int half  = lane >> 4;
  const int t   = blockIdx.x * (blockDim.x >> 5) + (threadIdx.x >> 5);
  const int mt  = t & 511;                   // 8192/16 = 512 row tiles
  const int ntg = t >> 9;                    // 128 groups of 64 columns
  const int m0 = mt << 4, n0 = ntg << 6;

  const unsigned short* arow = Zb + (size_t)(m0 + laneL) * KLAT;
  const unsigned short* brow = Zb + (size_t)(n0 + laneL) * KLAT;

  v8f acc0 = {}, acc1 = {}, acc2 = {}, acc3 = {};
#pragma unroll
  for (int kk = 0; kk < KLAT; kk += 32) {
    VgaeBfFrag a, b0, b1, b2, b3;
    a.q[0] = *(const v4u*)(arow + kk + (half << 4));
    a.q[1] = *(const v4u*)(arow + kk + 8 + (half << 4));
    const unsigned short* bp = brow + kk + (half << 4);
    b0.q[0] = *(const v4u*)(bp);                  b0.q[1] = *(const v4u*)(bp + 8);
    b1.q[0] = *(const v4u*)(bp + 16 * KLAT);      b1.q[1] = *(const v4u*)(bp + 16 * KLAT + 8);
    b2.q[0] = *(const v4u*)(bp + 32 * KLAT);      b2.q[1] = *(const v4u*)(bp + 32 * KLAT + 8);
    b3.q[0] = *(const v4u*)(bp + 48 * KLAT);      b3.q[1] = *(const v4u*)(bp + 48 * KLAT + 8);
    acc0 = __builtin_amdgcn_wmma_f32_16x16x32_bf16(false, a.v, false, b0.v,
                                                   (short)0, acc0, false, false);
    acc1 = __builtin_amdgcn_wmma_f32_16x16x32_bf16(false, a.v, false, b1.v,
                                                   (short)0, acc1, false, false);
    acc2 = __builtin_amdgcn_wmma_f32_16x16x32_bf16(false, a.v, false, b2.v,
                                                   (short)0, acc2, false, false);
    acc3 = __builtin_amdgcn_wmma_f32_16x16x32_bf16(false, a.v, false, b3.v,
                                                   (short)0, acc3, false, false);
  }
  vgae_store_sig(acc0, adj, m0, n0,      laneL, half);
  vgae_store_sig(acc1, adj, m0, n0 + 16, laneL, half);
  vgae_store_sig(acc2, adj, m0, n0 + 32, laneL, half);
  vgae_store_sig(acc3, adj, m0, n0 + 48, laneL, half);
}

// ------------------------- GCN aggregation kernels -------------------------

__global__ void vgae_agg_init(const float* __restrict__ H, const float* __restrict__ dinv,
                              const float* __restrict__ bias, float* __restrict__ out, int F) {
  long i = (long)blockIdx.x * blockDim.x + threadIdx.x;   // N*F threads
  int node = (int)(i / F);
  int f    = (int)(i - (long)node * F);
  float di = dinv[node];
  out[i] = H[i] * di * di + bias[f];                      // self loop + bias
}

__global__ void vgae_agg_edges(const float* __restrict__ H, const float* __restrict__ dinv,
                               const int* __restrict__ src, const int* __restrict__ dst,
                               float* __restrict__ out, int F) {
  long i = (long)blockIdx.x * blockDim.x + threadIdx.x;   // E*F threads
  int e = (int)(i / F);
  int f = (int)(i - (long)e * F);
  int s = src[e], d = dst[e];
  atomicAdd(&out[(size_t)d * F + f], H[(size_t)s * F + f] * dinv[s] * dinv[d]);
}

__global__ void vgae_relu(float* __restrict__ p, long n) {
  long i = (long)blockIdx.x * blockDim.x + threadIdx.x;
  if (i < n) p[i] = fmaxf(p[i], 0.0f);
}

// z = eps * exp(logstd) + mu, emitted directly as packed bf16 for z@z^T.
__global__ void vgae_reparam_bf16(const float* __restrict__ eps, const float* __restrict__ mu,
                                  const float* __restrict__ ls,
                                  unsigned short* __restrict__ zb, long n) {
  long i = (long)blockIdx.x * blockDim.x + threadIdx.x;
  if (i < n) zb[i] = vgae_bf16_rne(eps[i] * expf(ls[i]) + mu[i]);
}

// ----------------------------- host launcher -------------------------------

extern "C" void kernel_launch(void* const* d_in, const int* in_sizes, int n_in,
                              void* d_out, int out_size, void* d_ws, size_t ws_size,
                              hipStream_t stream) {
  (void)in_sizes; (void)n_in; (void)out_size; (void)ws_size;

  const float* x   = (const float*)d_in[0];
  const int*   ei  = (const int*)d_in[1];      // JAX x64-off downcasts int64->int32
  const float* eps = (const float*)d_in[2];
  const float* W1  = (const float*)d_in[3];
  const float* b1  = (const float*)d_in[4];
  const float* Wmu = (const float*)d_in[5];
  const float* bmu = (const float*)d_in[6];
  const float* Wls = (const float*)d_in[7];
  const float* bls = (const float*)d_in[8];
  const int* src = ei;
  const int* dst = ei + KE;

  float* adj = (float*)d_out;                            // [8192 x 8192]
  float* mu  = adj + (size_t)KN * KN;                    // [8192 x 128]
  float* ls  = mu + (size_t)KN * KLAT;                   // [8192 x 128]

  // workspace layout (peak ~16 MB)
  char* w = (char*)d_ws;
  float* dinv = (float*)w;            w += (size_t)KN * sizeof(float);
  float* H1   = (float*)w;            w += (size_t)KN * KHID * sizeof(float);
  float* h    = (float*)w;            /* KN*KHID floats */
  float* Hmu  = H1;                                  // H1 dead after layer-1 agg
  float* Hls  = H1 + (size_t)KN * KLAT;
  unsigned short* zb = (unsigned short*)h;           // h dead after mu/ls GEMMs

  const int T = 256;

  // degrees -> D^{-1/2}
  vgae_fill1<<<(KN + T - 1) / T, T, 0, stream>>>(dinv, KN);
  vgae_deg_edges<<<KE / T, T, 0, stream>>>(dst, dinv);
  vgae_deg_to_dinv<<<(KN + T - 1) / T, T, 0, stream>>>(dinv, KN);

  // layer 1: H1 = x @ W1 ; h = relu(agg(H1) + b1)
  vgae_gemm_bf16x3<<<(KN / 16) * (KHID / 16) / 8, T, 0, stream>>>(x, W1, H1, KN, KHID, KIN);
  vgae_agg_init<<<(long)KN * KHID / T, T, 0, stream>>>(H1, dinv, b1, h, KHID);
  vgae_agg_edges<<<(long)KE * KHID / T, T, 0, stream>>>(H1, dinv, src, dst, h, KHID);
  vgae_relu<<<(long)KN * KHID / T, T, 0, stream>>>(h, (long)KN * KHID);

  // mu / logstd heads
  vgae_gemm_bf16x3<<<(KN / 16) * (KLAT / 16) / 8, T, 0, stream>>>(h, Wmu, Hmu, KN, KLAT, KHID);
  vgae_gemm_bf16x3<<<(KN / 16) * (KLAT / 16) / 8, T, 0, stream>>>(h, Wls, Hls, KN, KLAT, KHID);
  vgae_agg_init<<<(long)KN * KLAT / T, T, 0, stream>>>(Hmu, dinv, bmu, mu, KLAT);
  vgae_agg_edges<<<(long)KE * KLAT / T, T, 0, stream>>>(Hmu, dinv, src, dst, mu, KLAT);
  vgae_agg_init<<<(long)KN * KLAT / T, T, 0, stream>>>(Hls, dinv, bls, ls, KLAT);
  vgae_agg_edges<<<(long)KE * KLAT / T, T, 0, stream>>>(Hls, dinv, src, dst, ls, KLAT);

  // z (bf16 packed) ; adj = sigmoid(z z^T) with 16x64 wave tiles
  vgae_reparam_bf16<<<(long)KN * KLAT / T, T, 0, stream>>>(eps, mu, ls, zb, (long)KN * KLAT);
  vgae_zzt_sigmoid<<<(KN / 16) * (KN / 64) / 8, T, 0, stream>>>(zb, adj);
}
